// Cos_RootHist_GLM_71683004170387
// MI455X (gfx1250) — matrix operand order repaired and checked by reading, not compiled, not run on hardware
//
#include <hip/hip_runtime.h>
#include <hip/hip_bf16.h>

// Problem constants (fixed by the reference).
#define T_DATA 20000
#define SUB_NO 15
#define N_E    2000
#define N_I    500
#define T_NO   200
#define NB     19

typedef __attribute__((ext_vector_type(16))) _Float16 v16h;
typedef __attribute__((ext_vector_type(8)))  float    v8f;
typedef __attribute__((ext_vector_type(4)))  float    f4;

__device__ __forceinline__ float softplus_f(float x) {
  // jax.nn.softplus: max(x,0) + log1p(exp(-|x|))
  return fmaxf(x, 0.f) + log1pf(expf(-fabsf(x)));
}

// ---------------------------------------------------------------------------
// Kernel A: e_kern/i_kern/hist_kern = W @ cos_basis; also emits out_filters.
// ---------------------------------------------------------------------------
__global__ __launch_bounds__(256) void filters_kernel(
    const float* __restrict__ W_syn, const float* __restrict__ W_hist,
    const float* __restrict__ CB, float* __restrict__ eK,
    float* __restrict__ iK, float* __restrict__ hK,
    float* __restrict__ out_filters) {
  int i = blockIdx.x * 256 + threadIdx.x;
  if (i >= 31 * T_NO) return;
  int r = i / T_NO, tau = i - r * T_NO;
  float acc = 0.f;
  if (r < SUB_NO) {
#pragma unroll
    for (int b = 0; b < NB; ++b)
      acc = fmaf(W_syn[(r * NB + b) * 2 + 0], CB[b * T_NO + tau], acc);
    eK[r * T_NO + tau] = acc;
  } else if (r < 2 * SUB_NO) {
    int s = r - SUB_NO;
#pragma unroll
    for (int b = 0; b < NB; ++b)
      acc = fmaf(W_syn[(s * NB + b) * 2 + 1], CB[b * T_NO + tau], acc);
    iK[s * T_NO + tau] = acc;
  } else {
#pragma unroll
    for (int b = 0; b < NB; ++b)
      acc = fmaf(W_hist[b], CB[b * T_NO + tau], acc);
    hK[tau] = acc;
  }
  out_filters[i] = acc;
}

// ---------------------------------------------------------------------------
// Kernel A2: zero-padded B matrix: Bpad[16][Kpad] = Csyn[15][K] (rest zero).
// Lets the GEMM run guard-free: garbage A values at k>=K hit zero B columns.
// ---------------------------------------------------------------------------
__global__ __launch_bounds__(256) void pad_csyn_kernel(
    const float* __restrict__ Csyn, float* __restrict__ Bpad, int K, int Kpad) {
  int i = blockIdx.x * 256 + threadIdx.x;
  if (i >= 16 * Kpad) return;
  int r = i / Kpad, k = i - r * Kpad;
  Bpad[i] = (r < SUB_NO && k < K) ? Csyn[r * K + k] : 0.f;
}

// ---------------------------------------------------------------------------
// Kernel B: syn = S @ Csynᵀ via v_wmma_f32_16x16x32_f16.
// Guard-free inner loop:
//   - A row index clamped to T_DATA-1 (dup rows computed, never stored)
//   - A float4 index clamped to last aligned quad of the row (k>=K values
//     multiply zero-padded B columns -> contribute 0; spikes are finite)
//   - B loads unconditional from the zero-padded 16 x Kpad buffer
// A is 160MB single-use stream -> non-temporal loads + prefetch 2 tiles ahead;
// output write-once -> non-temporal stores, wave-uniform fast-path epilogue.
// One wave = one 16x16 tile; 8 waves/block = 128 rows.
// ---------------------------------------------------------------------------
__global__ __launch_bounds__(256) void spike_gemm_kernel(
    const float* __restrict__ S, const float* __restrict__ Bpad,
    float* __restrict__ out, int K, int Kpad) {
  const int lane = threadIdx.x & 31;
  const int wave = threadIdx.x >> 5;
  const int hi   = lane >> 4;      // lane half selects K sub-block
  const int mn   = lane & 15;      // A: row M ; B: col N
  const int t0   = blockIdx.x * 128 + wave * 16;
  int trow = t0 + mn;
  if (trow > T_DATA - 1) trow = T_DATA - 1;            // clamp: loads always valid
  const f4* __restrict__ pa = (const f4*)(S + (size_t)trow * (size_t)K);
  const f4* __restrict__ pb = (const f4*)(Bpad + (size_t)mn * (size_t)Kpad);
  const int f4max = (K >> 2) - 1;                      // K % 4 == 0

  v8f acc = {};
  for (int k0 = 0; k0 < Kpad; k0 += 32) {
    // A fragment: 16x32 f16. lane&15 = M; elems 0..7 -> K=ka..ka+7,
    // elems 8..15 -> K=ka+16..ka+23, ka = k0 + 8*(lane>=16).
    const int ka4 = (k0 >> 2) + hi * 2;
    f4 x0 = __builtin_nontemporal_load(&pa[min(ka4 + 0, f4max)]);
    f4 x1 = __builtin_nontemporal_load(&pa[min(ka4 + 1, f4max)]);
    f4 x2 = __builtin_nontemporal_load(&pa[min(ka4 + 4, f4max)]);
    f4 x3 = __builtin_nontemporal_load(&pa[min(ka4 + 5, f4max)]);
    // Prefetch the A stream two k-tiles ahead (global_prefetch_b8).
    __builtin_prefetch(&pa[min(ka4 + 16, f4max)], 0, 1);
    // B fragment: 32x16 f16. lane&15 = N; elems 0..7 -> K=kb..kb+7,
    // elems 8..15 -> K=kb+16..kb+23, kb = k0 + 8*(lane>=16). Padded buffer -> always in bounds.
    const int kb4 = (k0 >> 2) + hi * 2;
    f4 y0 = pb[kb4 + 0], y1 = pb[kb4 + 1], y2 = pb[kb4 + 4], y3 = pb[kb4 + 5];

    v16h a, b;
    a[0]=(_Float16)x0[0];  a[1]=(_Float16)x0[1];  a[2]=(_Float16)x0[2];  a[3]=(_Float16)x0[3];
    a[4]=(_Float16)x1[0];  a[5]=(_Float16)x1[1];  a[6]=(_Float16)x1[2];  a[7]=(_Float16)x1[3];
    a[8]=(_Float16)x2[0];  a[9]=(_Float16)x2[1];  a[10]=(_Float16)x2[2]; a[11]=(_Float16)x2[3];
    a[12]=(_Float16)x3[0]; a[13]=(_Float16)x3[1]; a[14]=(_Float16)x3[2]; a[15]=(_Float16)x3[3];
    b[0]=(_Float16)y0[0];  b[1]=(_Float16)y0[1];  b[2]=(_Float16)y0[2];  b[3]=(_Float16)y0[3];
    b[4]=(_Float16)y1[0];  b[5]=(_Float16)y1[1];  b[6]=(_Float16)y1[2];  b[7]=(_Float16)y1[3];
    b[8]=(_Float16)y2[0];  b[9]=(_Float16)y2[1];  b[10]=(_Float16)y2[2]; b[11]=(_Float16)y2[3];
    b[12]=(_Float16)y3[0]; b[13]=(_Float16)y3[1]; b[14]=(_Float16)y3[2]; b[15]=(_Float16)y3[3];

    acc = __builtin_amdgcn_wmma_f32_16x16x32_f16(
        /*neg_a=*/false, a, /*neg_b=*/false, b,
        /*c_mod=*/(short)0, acc, /*reuse_a=*/false, /*reuse_b=*/false);
  }
  // C/D layout: VGPR r, lanes 0-15 -> M=r, lanes 16-31 -> M=8+r; N = lane&15.
  const int rbase = hi * 8;
  float* __restrict__ po = out + (size_t)(t0 + rbase) * 16 + mn;
  if (t0 + 16 <= T_DATA) {          // wave-uniform: full tile in bounds
#pragma unroll
    for (int r = 0; r < 8; ++r)
      __builtin_nontemporal_store(acc[r], po + (size_t)r * 16);
  } else {                          // tail tiles only
#pragma unroll
    for (int r = 0; r < 8; ++r)
      if (t0 + rbase + r < T_DATA)
        __builtin_nontemporal_store(acc[r], po + (size_t)r * 16);
  }
}

// ---------------------------------------------------------------------------
// Kernel C: depthwise causal conv (200 taps, 15 channels, e+i) + subunit tree.
// 256 timesteps per block; halo'd syn tiles staged in LDS (pitch 17 ->
// stride 17 coprime with 64 banks => conflict-free column access).
// ---------------------------------------------------------------------------
#define CT_TILE  256
#define CT_ROWS  (CT_TILE + T_NO - 1)   // 455
#define CT_PITCH 17

__global__ __launch_bounds__(256) void conv_tree_kernel(
    const float* __restrict__ syn_e, const float* __restrict__ syn_i,
    const float* __restrict__ eK, const float* __restrict__ iK,
    const float* __restrict__ C_den, const float* __restrict__ W_sub,
    const float* __restrict__ Theta,
    float* __restrict__ ns_out, float* __restrict__ b_out) {
  __shared__ float SE[CT_ROWS * CT_PITCH];
  __shared__ float SI[CT_ROWS * CT_PITCH];
  __shared__ float CD[SUB_NO * SUB_NO];
  __shared__ float W2[SUB_NO];
  __shared__ float TH[SUB_NO];
  const int tid = threadIdx.x;
  const int t0  = blockIdx.x * CT_TILE;

  for (int i = tid; i < CT_ROWS * 4; i += 256) {
    int row = i >> 2, c4 = i & 3;
    int g = t0 - (T_NO - 1) + row;
    f4 ve = {0.f, 0.f, 0.f, 0.f}, vi = {0.f, 0.f, 0.f, 0.f};
    if ((unsigned)g < (unsigned)T_DATA) {
      ve = ((const f4*)syn_e)[(size_t)g * 4 + c4];
      vi = ((const f4*)syn_i)[(size_t)g * 4 + c4];
    }
    int base = row * CT_PITCH + c4 * 4;
    SE[base+0]=ve[0]; SE[base+1]=ve[1]; SE[base+2]=ve[2]; SE[base+3]=ve[3];
    SI[base+0]=vi[0]; SI[base+1]=vi[1]; SI[base+2]=vi[2]; SI[base+3]=vi[3];
  }
  if (tid < SUB_NO * SUB_NO) CD[tid] = C_den[tid];
  if (tid < SUB_NO) { float w = W_sub[tid]; W2[tid] = w * w; TH[tid] = Theta[tid]; }
  __syncthreads();

  const int t = t0 + tid;
  if (t < T_DATA) {
    float syn[SUB_NO];
    for (int s = 0; s < SUB_NO; ++s) {
      const float* __restrict__ ek = eK + s * T_NO;
      const float* __restrict__ ik = iK + s * T_NO;
      float a = 0.f;
#pragma unroll 4
      for (int tau = 0; tau < T_NO; ++tau) {
        int row = tid + (T_NO - 1) - tau;           // global t - tau
        a = fmaf(ek[tau], SE[row * CT_PITCH + s], a);
        a = fmaf(ik[tau], SI[row * CT_PITCH + s], a);
      }
      syn[s] = a;
    }
    float ns[SUB_NO];
#pragma unroll
    for (int s = 0; s < SUB_NO; ++s) ns[s] = 0.f;
#pragma unroll
    for (int j = SUB_NO - 1; j >= 1; --j) {
      float leaf = 0.f;
#pragma unroll
      for (int c = 0; c < SUB_NO; ++c)
        leaf = fmaf(CD[j * SUB_NO + c] * W2[c], ns[c], leaf);
      ns[j] = softplus_f(syn[j] + leaf + TH[j]);
    }
    float leaf0 = 0.f;
#pragma unroll
    for (int c = 0; c < SUB_NO; ++c)
      leaf0 = fmaf(CD[c] * W2[c], ns[c], leaf0);
    b_out[t] = syn[0] + leaf0 + TH[0];
#pragma unroll
    for (int s = 0; s < SUB_NO; ++s)
      ns_out[(size_t)t * SUB_NO + s] = ns[s];      // ns[0] stays 0 (ref behavior)
  }
}

// ---------------------------------------------------------------------------
// Kernel D: sequential GLM scan. One wave32; 200 history taps split 7/lane in
// registers; y history in a 256-entry LDS ring; shfl_xor wave reduction.
// ---------------------------------------------------------------------------
__global__ __launch_bounds__(32) void scan_kernel(
    const float* __restrict__ b, const float* __restrict__ hK,
    const float* __restrict__ W_sub, const float* __restrict__ V_o,
    float* __restrict__ outV) {
  __shared__ float ring[256];
  const int lane = threadIdx.x;
  for (int i = lane; i < 256; i += 32) ring[i] = 0.f;
  float h[7];
#pragma unroll
  for (int j = 0; j < 7; ++j) {
    int tap = lane + 32 * j;                 // tap = tau-1
    h[j] = (tap < T_NO) ? hK[tap] : 0.f;
  }
  float w0 = W_sub[0]; w0 *= w0;
  const float vo = V_o[0];
  __syncthreads();
  for (int t = 0; t < T_DATA; ++t) {
    float dot = 0.f;
#pragma unroll
    for (int j = 0; j < 7; ++j) {
      int tap = lane + 32 * j;               // y[t - (tap+1)]; h[j]==0 masks OOB
      dot = fmaf(h[j], ring[(t - tap - 1) & 255], dot);
    }
#pragma unroll
    for (int off = 16; off >= 1; off >>= 1)
      dot += __shfl_xor(dot, off, 32);
    float y = softplus_f(b[t] + dot);
    if (lane == 0) {
      ring[t & 255] = y;
      outV[t] = fmaf(y, w0, vo);
    }
    __syncthreads();
  }
}

// ---------------------------------------------------------------------------
extern "C" void kernel_launch(void* const* d_in, const int* in_sizes, int n_in,
                              void* d_out, int out_size, void* d_ws, size_t ws_size,
                              hipStream_t stream) {
  (void)in_sizes; (void)n_in; (void)out_size; (void)ws_size;
  const float* S_e     = (const float*)d_in[0];
  const float* S_i     = (const float*)d_in[1];
  const float* C_den   = (const float*)d_in[2];
  const float* C_syn_e = (const float*)d_in[3];
  const float* C_syn_i = (const float*)d_in[4];
  const float* CB      = (const float*)d_in[5];
  const float* W_syn   = (const float*)d_in[6];
  const float* W_sub   = (const float*)d_in[7];
  const float* V_o     = (const float*)d_in[8];
  const float* Theta   = (const float*)d_in[9];
  const float* W_hist  = (const float*)d_in[10];

  float* out         = (float*)d_out;
  float* final_V     = out;                                    // 20000
  float* ns_out      = out + T_DATA;                           // 20000*15
  float* out_filters = out + T_DATA + (size_t)T_DATA * SUB_NO; // 31*200

  const int KPAD_E = (N_E + 31) & ~31;   // 2016
  const int KPAD_I = (N_I + 31) & ~31;   // 512

  float* ws     = (float*)d_ws;
  float* syn_e  = ws;                                // 20000*16
  float* syn_i  = syn_e + (size_t)T_DATA * 16;       // 20000*16
  float* b_ws   = syn_i + (size_t)T_DATA * 16;       // 20000
  float* eK     = b_ws + T_DATA;                     // 15*200
  float* iK     = eK + SUB_NO * T_NO;                // 15*200
  float* hK     = iK + SUB_NO * T_NO;                // 200
  float* Bpad_e = hK + T_NO;                         // 16*KPAD_E
  float* Bpad_i = Bpad_e + 16 * KPAD_E;              // 16*KPAD_I

  filters_kernel<<<(31 * T_NO + 255) / 256, 256, 0, stream>>>(
      W_syn, W_hist, CB, eK, iK, hK, out_filters);
  pad_csyn_kernel<<<(16 * KPAD_E + 255) / 256, 256, 0, stream>>>(
      C_syn_e, Bpad_e, N_E, KPAD_E);
  pad_csyn_kernel<<<(16 * KPAD_I + 255) / 256, 256, 0, stream>>>(
      C_syn_i, Bpad_i, N_I, KPAD_I);
  spike_gemm_kernel<<<(T_DATA + 127) / 128, 256, 0, stream>>>(
      S_e, Bpad_e, syn_e, N_E, KPAD_E);
  spike_gemm_kernel<<<(T_DATA + 127) / 128, 256, 0, stream>>>(
      S_i, Bpad_i, syn_i, N_I, KPAD_I);
  conv_tree_kernel<<<(T_DATA + CT_TILE - 1) / CT_TILE, 256, 0, stream>>>(
      syn_e, syn_i, eK, iK, C_den, W_sub, Theta, ns_out, b_ws);
  scan_kernel<<<1, 32, 0, stream>>>(b_ws, hK, W_sub, V_o, final_V);
}
